// KANLinear_1340029797083
// MI455X (gfx1250) — compile-verified, matrix operand order checked by reading, not run on hardware
//
#include <hip/hip_runtime.h>
#include <hip/hip_bf16.h>

typedef _Float16 v16h __attribute__((ext_vector_type(16)));
typedef _Float16 v8h  __attribute__((ext_vector_type(8)));
typedef float    v8f  __attribute__((ext_vector_type(8)));
typedef unsigned int u32x4 __attribute__((ext_vector_type(4)));
typedef int          i32x4 __attribute__((ext_vector_type(4)));
typedef int          i32x8 __attribute__((ext_vector_type(8)));

constexpr int Bn   = 2048;  // batch (GEMM M)
constexpr int Dn   = 256;   // input dims
constexpr int OUTn = 256;   // outputs (GEMM N)
constexpr int Kn   = 16;    // knots
constexpr int KD   = Dn * Kn;           // GEMM inner dim = 4096
constexpr float GSTEP = 2.0f / 15.0f;   // grid spacing

#if defined(__gfx1250__) && __has_builtin(__builtin_amdgcn_tensor_load_to_lds) && \
    __has_builtin(__builtin_amdgcn_s_wait_tensorcnt)
#define USE_TDM 1
#else
#define USE_TDM 0
#endif

// ---------------------------------------------------------------------------
// Kernel 1: build sparse-interp coefficient matrix C (Bn x KD) in f16.
// Per (b,d): c[left] = 1-w, c[left+1] = w, rest 0.
// ---------------------------------------------------------------------------
__global__ void kan_coeff(const float* __restrict__ x, _Float16* __restrict__ A) {
    int idx = blockIdx.x * blockDim.x + threadIdx.x;      // 0 .. Bn*Dn-1
    float xv = x[idx];
    float xc = fminf(fmaxf(xv, -1.0f), 1.0f);
    float t = (xc + 1.0f) * 7.5f;
    int right = (int)ceilf(t);
    int left = right - 1;
    left = left < 0 ? 0 : (left > Kn - 2 ? Kn - 2 : left);
    int rgt = left + 1;
    float g0 = -1.0f + (float)left * GSTEP;
    float w = (xc - g0) * (1.0f / (GSTEP + 1e-12f));
    float om = 1.0f - w;

    v8h lo = {};
    v8h hi = {};
#pragma unroll
    for (int k = 0; k < 8; ++k) {
        float cv = (k == left) ? om : ((k == rgt) ? w : 0.0f);
        lo[k] = (_Float16)cv;
    }
#pragma unroll
    for (int k = 8; k < 16; ++k) {
        float cv = (k == left) ? om : ((k == rgt) ? w : 0.0f);
        hi[k - 8] = (_Float16)cv;
    }
    v8h* row = (v8h*)(A + (size_t)idx * Kn);   // 32-byte aligned
    row[0] = lo;
    row[1] = hi;
}

// ---------------------------------------------------------------------------
// Kernel 2: B^T (OUTn x KD, row-major over o) in f16 with skip folded in:
//   BT[o][d*16+k] = values[o,d,k] + skip_w[o,d]*grid[k]
// ---------------------------------------------------------------------------
__global__ void kan_bmat(const float* __restrict__ values,
                         const float* __restrict__ skip_w,
                         _Float16* __restrict__ BT) {
    int idx = blockIdx.x * blockDim.x + threadIdx.x;      // 0 .. OUTn*KD-1
    int o = idx >> 12;
    int r = idx & (KD - 1);
    int d = r >> 4;
    int k = r & 15;
    float g = -1.0f + (float)k * GSTEP;
    float v = values[idx] + skip_w[o * Dn + d] * g;
    BT[(size_t)o * KD + r] = (_Float16)v;
}

// ---------------------------------------------------------------------------
// TDM helper: 2D tensor_load_to_lds of a (tile_d1 rows x 32 halves) tile.
// data_size = 2 bytes; LDS padding: +16 B after every 64 B  -> 80 B row
// stride in LDS (matches LDS_STRIDE = 40 halves below).
// ---------------------------------------------------------------------------
#if USE_TDM
__device__ static inline void tdm_load_2d(unsigned long long gaddr, unsigned ldsaddr,
                                          unsigned tensor_d0, unsigned tensor_d1,
                                          unsigned tile_d0, unsigned tile_d1) {
    u32x4 g0;
    g0[0] = 1u;                                          // count=1 (valid user D#)
    g0[1] = ldsaddr;                                     // LDS byte address
    g0[2] = (unsigned)(gaddr & 0xFFFFFFFFull);           // global_addr[31:0]
    g0[3] = (unsigned)((gaddr >> 32) & 0x1FFFFFFull)     // global_addr[56:32]
            | (2u << 30);                                // type = 2 ("image")
    i32x8 g1;
    g1[0] = (int)((1u << 16)      // data_size = 1 -> 2 bytes
                | (1u << 20)      // pad_enable
                | (3u << 22)      // pad_interval: every 16 DWORDs (64 B)
                | (3u << 25));    // pad_amount: 4 DWORDs (16 B)
    g1[1] = (int)((tensor_d0 & 0xFFFFu) << 16);                          // dim0[15:0]
    g1[2] = (int)(((tensor_d0 >> 16) & 0xFFFFu) | ((tensor_d1 & 0xFFFFu) << 16));
    g1[3] = (int)(((tensor_d1 >> 16) & 0xFFFFu) | ((tile_d0 & 0xFFFFu) << 16));
    g1[4] = (int)(tile_d1 & 0xFFFFu);                                    // tile_dim1
    g1[5] = (int)tensor_d0;        // tensor_dim0_stride[31:0] (= row pitch)
    g1[6] = 0;                     // stride high bits / dim1_stride (2D: unused)
    g1[7] = 0;
    i32x4 z4 = {0, 0, 0, 0};
#if __clang_major__ >= 23
    i32x8 z8 = {0, 0, 0, 0, 0, 0, 0, 0};
    __builtin_amdgcn_tensor_load_to_lds(g0, g1, z4, z4, z8, 0);
#else
    __builtin_amdgcn_tensor_load_to_lds(g0, g1, z4, z4, 0);
#endif
}
#endif

// ---------------------------------------------------------------------------
// Kernel 3: GEMM out = C * BT^T + skip_b via v_wmma_f32_16x16x32_f16.
// 256 threads = 8 wave32 waves; 128(M) x 64(N) tile per block; each wave a
// 32x32 sub-tile (2x2 WMMA accumulators). K in steps of 32 through LDS.
// Tile fill: TDM (tensor_load_to_lds + s_wait_tensorcnt) when available,
// else global_load_b128 -> ds_store_b128 staging.
// ---------------------------------------------------------------------------
constexpr int LDS_STRIDE = 40;   // 32 halves + 8 pad = 80 B rows (16B aligned)

__global__ __launch_bounds__(256)
void kan_gemm(const _Float16* __restrict__ A, const _Float16* __restrict__ BT,
              const float* __restrict__ skip_b, float* __restrict__ out) {
    __shared__ alignas(16) _Float16 sA[128 * LDS_STRIDE];
    __shared__ alignas(16) _Float16 sB[64 * LDS_STRIDE];

    const int tid   = threadIdx.x;
    const int wid   = tid >> 5;          // wave32
    const int lane  = tid & 31;
    const int waveM = wid >> 1;          // 0..3  -> M offset waveM*32
    const int waveN = wid & 1;           // 0..1  -> N offset waveN*32
    const int mlane = lane & 15;
    const int half  = lane >> 4;         // lane half selects K sub-range

    const int blockM = blockIdx.x;       // 0..15
    const int blockN = blockIdx.y;       // 0..3

    const _Float16* Ablk = A  + (size_t)(blockM * 128) * KD;
    const _Float16* Bblk = BT + (size_t)(blockN * 64) * KD;

    v8f acc[2][2] = {};

#if USE_TDM
    const unsigned ldsA_off = (unsigned)(uintptr_t)(&sA[0]);
    const unsigned ldsB_off = (unsigned)(uintptr_t)(&sB[0]);
    const unsigned long long gA = (unsigned long long)(uintptr_t)Ablk;
    const unsigned long long gB = (unsigned long long)(uintptr_t)Bblk;
#else
    const int rowA0 = tid >> 2;          // global-load chunk assignments
    const int rowA1 = (tid + 256) >> 2;
    const int cA    = tid & 3;
    const int rowB  = tid >> 2;
    const int cB    = tid & 3;
#endif

    for (int kb = 0; kb < KD / 32; ++kb) {
        const int k0 = kb * 32;

#if USE_TDM
        __syncthreads();                 // all waves done reading previous tile
        if (wid == 0) {
            // A tile: 128 rows x 32 halves from (2048 x 4096) tensor
            tdm_load_2d(gA + (unsigned long long)k0 * 2, ldsA_off,
                        (unsigned)KD, (unsigned)Bn, 32u, 128u);
            // B tile: 64 rows x 32 halves from (256 x 4096) tensor
            tdm_load_2d(gB + (unsigned long long)k0 * 2, ldsB_off,
                        (unsigned)KD, (unsigned)OUTn, 32u, 64u);
            __builtin_amdgcn_s_wait_tensorcnt(0);
        }
        __syncthreads();                 // DMA'd tiles visible to all waves
#else
        v8h a0 = *(const v8h*)(Ablk + (size_t)rowA0 * KD + k0 + cA * 8);
        v8h a1 = *(const v8h*)(Ablk + (size_t)rowA1 * KD + k0 + cA * 8);
        v8h b0 = *(const v8h*)(Bblk + (size_t)rowB * KD + k0 + cB * 8);
        if (kb + 1 < KD / 32) {
            __builtin_prefetch(Ablk + (size_t)rowA0 * KD + k0 + 32 + cA * 8, 0, 0);
        }
        __syncthreads();
        *(v8h*)(sA + rowA0 * LDS_STRIDE + cA * 8) = a0;
        *(v8h*)(sA + rowA1 * LDS_STRIDE + cA * 8) = a1;
        *(v8h*)(sB + rowB  * LDS_STRIDE + cB * 8) = b0;
        __syncthreads();
#endif

        // Fragments per ISA 16-bit A layout:
        //   VGPR0-3 : K = half*8 .. half*8+7     (elements 0..7)
        //   VGPR4-7 : K = 16+half*8 .. +7        (elements 8..15)
        v16h afrag[2], bfrag[2];
#pragma unroll
        for (int mt = 0; mt < 2; ++mt) {
            const _Float16* rp = sA + (waveM * 32 + mt * 16 + mlane) * LDS_STRIDE;
            v8h lo = *(const v8h*)(rp + half * 8);
            v8h hi = *(const v8h*)(rp + 16 + half * 8);
            afrag[mt] = __builtin_shufflevector(lo, hi, 0, 1, 2, 3, 4, 5, 6, 7,
                                                8, 9, 10, 11, 12, 13, 14, 15);
        }
#pragma unroll
        for (int nt = 0; nt < 2; ++nt) {
            const _Float16* rp = sB + (waveN * 32 + nt * 16 + mlane) * LDS_STRIDE;
            v8h lo = *(const v8h*)(rp + half * 8);
            v8h hi = *(const v8h*)(rp + 16 + half * 8);
            bfrag[nt] = __builtin_shufflevector(lo, hi, 0, 1, 2, 3, 4, 5, 6, 7,
                                                8, 9, 10, 11, 12, 13, 14, 15);
        }

#pragma unroll
        for (int mt = 0; mt < 2; ++mt)
#pragma unroll
            for (int nt = 0; nt < 2; ++nt)
                acc[mt][nt] = __builtin_amdgcn_wmma_f32_16x16x32_f16(
                    false, afrag[mt], false, bfrag[nt],
                    (short)0, acc[mt][nt], false, false);
    }

    // Epilogue: C/D layout -> VGPR r holds M=r (lanes 0-15) / M=8+r (lanes 16-31),
    // N = lane&15. Add skip_b and store f32.
#pragma unroll
    for (int nt = 0; nt < 2; ++nt) {
        const int col = blockN * 64 + waveN * 32 + nt * 16 + mlane;
        const float bias = skip_b[col];
#pragma unroll
        for (int mt = 0; mt < 2; ++mt) {
            const int rbase = blockM * 128 + waveM * 32 + mt * 16 + half * 8;
#pragma unroll
            for (int r = 0; r < 8; ++r) {
                out[(size_t)(rbase + r) * OUTn + col] = acc[mt][nt][r] + bias;
            }
        }
    }
}

// ---------------------------------------------------------------------------
extern "C" void kernel_launch(void* const* d_in, const int* in_sizes, int n_in,
                              void* d_out, int out_size, void* d_ws, size_t ws_size,
                              hipStream_t stream) {
    const float* x      = (const float*)d_in[0];   // (B, D)
    const float* values = (const float*)d_in[1];   // (OUT, D, K)
    const float* skip_w = (const float*)d_in[2];   // (OUT, D)
    const float* skip_b = (const float*)d_in[3];   // (OUT,)
    float* out = (float*)d_out;                    // (B, OUT)

    _Float16* Aws  = (_Float16*)d_ws;                                    // 16 MB
    _Float16* BTws = (_Float16*)((char*)d_ws + (size_t)Bn * KD * 2);     // +2 MB

    kan_coeff<<<(Bn * Dn) / 256, 256, 0, stream>>>(x, Aws);
    kan_bmat<<<(OUTn * KD) / 256, 256, 0, stream>>>(values, skip_w, BTws);
    kan_gemm<<<dim3(Bn / 128, OUTn / 64), 256, 0, stream>>>(Aws, BTws, skip_b, out);
}
